// AttentionBlock_16904991277668
// MI455X (gfx1250) — matrix operand order repaired and checked by reading, not compiled, not run on hardware
//
#include <hip/hip_runtime.h>
#include <hip/hip_fp16.h>

typedef __attribute__((ext_vector_type(16))) _Float16 v16h;
typedef __attribute__((ext_vector_type(8)))  _Float16 v8h;
typedef __attribute__((ext_vector_type(8)))  float    v8f;

constexpr int Bn  = 4;
constexpr int Sn  = 2048;
constexpr int Dn  = 768;
constexpr int Hn  = 12;
constexpr int DHn = 64;
constexpr int Mn  = Bn * Sn;          // 8192 tokens
constexpr int MT  = Mn / 16;          // 512 row tiles
constexpr int NT  = Dn / 16;          // 48 col tiles

// ---------------------------------------------------------------------------
// WMMA helpers (ISA 7.12.2 VGPR layouts)
// ---------------------------------------------------------------------------
__device__ inline v8f wmma_f16(v16h a, v16h b, v8f c) {
  return __builtin_amdgcn_wmma_f32_16x16x32_f16(
      /*neg_a=*/false, a, /*neg_b=*/false, b,
      /*c_mod=*/(short)0, c, /*reuse_a=*/false, /*reuse_b=*/false);
}

// A-matrix 16x32 f16: lane L holds row (L&15); lanes 0-15 hold K={0..7,16..23},
// lanes 16-31 hold K={8..15,24..31}.  Two 16-byte loads per lane.
__device__ inline v16h load_a_frag(const _Float16* base, int ld) {
  const int lane = threadIdx.x & 31;
  const int row  = lane & 15;
  const int kb   = (lane < 16) ? 0 : 8;
  const _Float16* p = base + (size_t)row * ld + kb;
  union { v16h v; v8h h[2]; } u;
  u.h[0] = *(const v8h*)(p);
  u.h[1] = *(const v8h*)(p + 16);
  return u.v;
}

// B-matrix 32x16 f16: lane L holds column (L&15); lanes 0-15 hold K={0..7,16..23},
// lanes 16-31 hold K={8..15,24..31}.  Two 16-byte loads per lane.
__device__ inline v16h load_b_frag(const _Float16* base, int ld) {
  const int lane = threadIdx.x & 31;
  const int col  = lane & 15;
  const int kb   = (lane < 16) ? 0 : 8;
  const _Float16* p = base + (size_t)col * ld + kb;
  union { v16h v; v8h h[2]; } u;
  u.h[0] = *(const v8h*)(p);
  u.h[1] = *(const v8h*)(p + 16);
  return u.v;
}

// 16-lane (half-wave) reductions for softmax rows
__device__ inline float hmax16(float v) {
  v = fmaxf(v, __shfl_xor(v, 1, 32));
  v = fmaxf(v, __shfl_xor(v, 2, 32));
  v = fmaxf(v, __shfl_xor(v, 4, 32));
  v = fmaxf(v, __shfl_xor(v, 8, 32));
  return v;
}
__device__ inline float hsum16(float v) {
  v += __shfl_xor(v, 1, 32);
  v += __shfl_xor(v, 2, 32);
  v += __shfl_xor(v, 4, 32);
  v += __shfl_xor(v, 8, 32);
  return v;
}

// ---------------------------------------------------------------------------
// Stage 0: fp32 -> fp16 conversion
// ---------------------------------------------------------------------------
__global__ void cvt_f32_f16_kernel(const float* __restrict__ src,
                                   _Float16* __restrict__ dst, int n) {
  int i = blockIdx.x * blockDim.x + threadIdx.x;
  if (i < n) dst[i] = (_Float16)src[i];
}

// ---------------------------------------------------------------------------
// Stage 1: fused QKV projection.  One wave computes a 64x16 output strip for
// all three weights.  Per k-step: 7 fragment loads up front (4 A + 3 B into
// distinct registers), one wait, then 12 back-to-back wmma.
// Q scaled by Dh^-0.5; V stored transposed [B,H,Dh,S].
// ---------------------------------------------------------------------------
__global__ __launch_bounds__(128) void qkv_kernel(
    const _Float16* __restrict__ xh,
    const _Float16* __restrict__ wq, const _Float16* __restrict__ wk,
    const _Float16* __restrict__ wv,
    _Float16* __restrict__ Qh, _Float16* __restrict__ Kh,
    _Float16* __restrict__ Vt) {
  const int MS = MT / 4;                         // 128 m-strips (64 rows each)
  const int wid = blockIdx.x * (blockDim.x >> 5) + (threadIdx.x >> 5);
  if (wid >= MS * NT) return;
  const int ms = wid / NT;
  const int nt = wid % NT;

  const _Float16* abase = xh + (size_t)(ms * 64) * Dn;
  const _Float16* wb[3] = { wq + (size_t)(nt * 16) * Dn,
                            wk + (size_t)(nt * 16) * Dn,
                            wv + (size_t)(nt * 16) * Dn };

  v8f acc[3][4];
#pragma unroll
  for (int w = 0; w < 3; ++w)
#pragma unroll
    for (int i = 0; i < 4; ++i) acc[w][i] = v8f{};

  for (int k = 0; k < Dn; k += 32) {
    v16h a[4], b[3];
#pragma unroll
    for (int i = 0; i < 4; ++i) {
      a[i] = load_a_frag(abase + (size_t)i * 16 * Dn + k, Dn);
      __builtin_prefetch(abase + (size_t)i * 16 * Dn + k + 64, 0, 3);
    }
#pragma unroll
    for (int w = 0; w < 3; ++w) {
      b[w] = load_b_frag(wb[w] + k, Dn);
      __builtin_prefetch(wb[w] + k + 64, 0, 3);
    }
#pragma unroll
    for (int w = 0; w < 3; ++w)
#pragma unroll
      for (int i = 0; i < 4; ++i) acc[w][i] = wmma_f16(a[i], b[w], acc[w][i]);
  }

  const int lane = threadIdx.x & 31;
  const int col  = lane & 15;
  const int rb   = (lane < 16) ? 0 : 8;
  const int n    = nt * 16 + col;
  const int h    = n >> 6;          // head
  const int d    = n & 63;          // dim within head
#pragma unroll
  for (int i = 0; i < 4; ++i) {
#pragma unroll
    for (int r = 0; r < 8; ++r) {
      const int m = ms * 64 + i * 16 + rb + r;
      const int b2 = m >> 11;       // batch (S=2048)
      const int s = m & 2047;
      const size_t qi = (((size_t)(b2 * Hn + h)) * Sn + s) * DHn + d;
      Qh[qi] = (_Float16)(acc[0][i][r] * 0.125f);   // 1/sqrt(64)
      Kh[qi] = (_Float16)acc[1][i][r];
      Vt[(((size_t)(b2 * Hn + h)) * DHn + d) * Sn + s] = (_Float16)acc[2][i][r];
    }
  }
}

// ---------------------------------------------------------------------------
// Stage 2: causal flash attention.  One wave per (b, h, 32-row query strip):
// two stacked 16-row query tiles share every K/V fragment (16 wmma per
// 32-key step from 8 B-frag loads).  Online softmax; P staged through LDS.
// V loads hoisted ahead of the softmax to overlap VALU with memory latency.
// ---------------------------------------------------------------------------
__global__ __launch_bounds__(32) void attn_kernel(
    const _Float16* __restrict__ Qh, const _Float16* __restrict__ Kh,
    const _Float16* __restrict__ Vt, _Float16* __restrict__ Oh) {
  __shared__ __align__(16) _Float16 Pst[2][16 * 32];

  const int ST = Sn / 32;                 // 64 query strips per (b,h)
  const int st = blockIdx.x % ST;
  const int bh = blockIdx.x / ST;         // b*H + h
  const int ib = st * 32;                 // first query row of strip

  const _Float16* qbase = Qh + ((size_t)bh * Sn + ib) * DHn;
  const _Float16* kbase = Kh + (size_t)bh * Sn * DHn;
  const _Float16* vbase = Vt + (size_t)bh * DHn * Sn;

  const int lane = threadIdx.x & 31;
  const int colL = lane & 15;
  const int rb   = (lane < 16) ? 0 : 8;

  // Q strip (32 x 64) as 2 tiles x 2 k-chunks of A-fragments
  v16h qa[2][2];
#pragma unroll
  for (int t = 0; t < 2; ++t) {
    qa[t][0] = load_a_frag(qbase + (size_t)t * 16 * DHn + 0,  DHn);
    qa[t][1] = load_a_frag(qbase + (size_t)t * 16 * DHn + 32, DHn);
  }

  v8f o[2][4];
  float m_r[2][8], l_r[2][8];
#pragma unroll
  for (int t = 0; t < 2; ++t) {
#pragma unroll
    for (int n4 = 0; n4 < 4; ++n4) o[t][n4] = v8f{};
#pragma unroll
    for (int r = 0; r < 8; ++r) { m_r[t][r] = -1e30f; l_r[t][r] = 0.0f; }
  }

  const int nk = ib + 32;                 // keys needed (causal)
  for (int jb = 0; jb < nk; jb += 32) {
    // ---- S = Q @ K^T : 4 score fragments (2 q-tiles x 2 key tiles)
    v8f s[2][2];
    {
      v16h k0 = load_b_frag(kbase + (size_t)jb * DHn + 0,  DHn);
      v16h k1 = load_b_frag(kbase + (size_t)jb * DHn + 32, DHn);
#pragma unroll
      for (int t = 0; t < 2; ++t) {
        s[t][0] = wmma_f16(qa[t][0], k0, v8f{});
        s[t][0] = wmma_f16(qa[t][1], k1, s[t][0]);
      }
      k0 = load_b_frag(kbase + (size_t)(jb + 16) * DHn + 0,  DHn);
      k1 = load_b_frag(kbase + (size_t)(jb + 16) * DHn + 32, DHn);
#pragma unroll
      for (int t = 0; t < 2; ++t) {
        s[t][1] = wmma_f16(qa[t][0], k0, v8f{});
        s[t][1] = wmma_f16(qa[t][1], k1, s[t][1]);
      }
    }

    // ---- issue V fragment loads early (only depend on jb); prefetch next
    //      j-step's K and V panel lines while the softmax VALU chain runs
    v16h vb[4];
#pragma unroll
    for (int n4 = 0; n4 < 4; ++n4)
      vb[n4] = load_b_frag(vbase + (size_t)(n4 * 16) * Sn + jb, Sn);
    __builtin_prefetch(kbase + (size_t)(jb + 32 + lane) * DHn,      0, 3);
    __builtin_prefetch(kbase + (size_t)(jb + 32 + lane) * DHn + 32, 0, 3);
    __builtin_prefetch(vbase + (size_t)lane        * Sn + jb + 32,  0, 3);
    __builtin_prefetch(vbase + (size_t)(32 + lane) * Sn + jb + 32,  0, 3);

    // ---- causal mask + online softmax, then stage P tiles into LDS
#pragma unroll
    for (int t = 0; t < 2; ++t) {
#pragma unroll
      for (int r = 0; r < 8; ++r) {
        const int ig = ib + t * 16 + rb + r;
        if (jb + colL      > ig) s[t][0][r] = -1e30f;
        if (jb + 16 + colL > ig) s[t][1][r] = -1e30f;
        float tm   = hmax16(fmaxf(s[t][0][r], s[t][1][r]));
        float mnew = fmaxf(m_r[t][r], tm);
        float alpha = __expf(m_r[t][r] - mnew);
        float p0 = __expf(s[t][0][r] - mnew);
        float p1 = __expf(s[t][1][r] - mnew);
        l_r[t][r] = l_r[t][r] * alpha + hsum16(p0 + p1);
        m_r[t][r] = mnew;
#pragma unroll
        for (int n4 = 0; n4 < 4; ++n4) o[t][n4][r] *= alpha;
        const int row = rb + r;
        Pst[t][row * 32 + colL]      = (_Float16)p0;
        Pst[t][row * 32 + 16 + colL] = (_Float16)p1;
      }
    }
    asm volatile("s_wait_dscnt 0x0" ::: "memory");
    const v16h pa0 = load_a_frag(&Pst[0][0], 32);
    const v16h pa1 = load_a_frag(&Pst[1][0], 32);
    asm volatile("" ::: "memory");

    // ---- O += P @ V   (V^T layout: row d contiguous over keys)
#pragma unroll
    for (int n4 = 0; n4 < 4; ++n4) {
      o[0][n4] = wmma_f16(pa0, vb[n4], o[0][n4]);
      o[1][n4] = wmma_f16(pa1, vb[n4], o[1][n4]);
    }
  }

  // ---- normalize and store O in [B, S, H*Dh] f16 layout for final GEMM
  const int b = bh / Hn, h = bh % Hn;
#pragma unroll
  for (int t = 0; t < 2; ++t) {
#pragma unroll
    for (int r = 0; r < 8; ++r) {
      const float inv = 1.0f / l_r[t][r];
      const size_t row =
          (size_t)(b * Sn + ib + t * 16 + rb + r) * Dn + h * DHn + colL;
      Oh[row +  0] = (_Float16)(o[t][0][r] * inv);
      Oh[row + 16] = (_Float16)(o[t][1][r] * inv);
      Oh[row + 32] = (_Float16)(o[t][2][r] * inv);
      Oh[row + 48] = (_Float16)(o[t][3][r] * inv);
    }
  }
}

// ---------------------------------------------------------------------------
// Stage 3: output projection  out = Oh @ Wo^T + bo  (64x32 strip per wave).
// Per k-step: 6 fragment loads up front, one wait, 8 back-to-back wmma.
// ---------------------------------------------------------------------------
__global__ __launch_bounds__(128) void out_proj_kernel(
    const _Float16* __restrict__ Oh, const _Float16* __restrict__ woh,
    const float* __restrict__ bo, float* __restrict__ out) {
  const int MS  = MT / 4;                        // 128 m-strips
  const int NT2 = NT / 2;                        // 24 n-strips (32 cols)
  const int wid = blockIdx.x * (blockDim.x >> 5) + (threadIdx.x >> 5);
  if (wid >= MS * NT2) return;
  const int ms = wid / NT2;
  const int nt = wid % NT2;

  const _Float16* abase = Oh + (size_t)(ms * 64) * Dn;
  const _Float16* bbase = woh + (size_t)(nt * 32) * Dn;

  v8f acc[2][4];
#pragma unroll
  for (int j = 0; j < 2; ++j)
#pragma unroll
    for (int i = 0; i < 4; ++i) acc[j][i] = v8f{};

  for (int k = 0; k < Dn; k += 32) {
    v16h a[4], b[2];
#pragma unroll
    for (int i = 0; i < 4; ++i) {
      a[i] = load_a_frag(abase + (size_t)i * 16 * Dn + k, Dn);
      __builtin_prefetch(abase + (size_t)i * 16 * Dn + k + 64, 0, 3);
    }
#pragma unroll
    for (int j = 0; j < 2; ++j) {
      b[j] = load_b_frag(bbase + (size_t)j * 16 * Dn + k, Dn);
      __builtin_prefetch(bbase + (size_t)j * 16 * Dn + k + 64, 0, 3);
    }
#pragma unroll
    for (int j = 0; j < 2; ++j)
#pragma unroll
      for (int i = 0; i < 4; ++i) acc[j][i] = wmma_f16(a[i], b[j], acc[j][i]);
  }

  const int lane = threadIdx.x & 31;
  const int col  = lane & 15;
  const int rb   = (lane < 16) ? 0 : 8;
#pragma unroll
  for (int j = 0; j < 2; ++j) {
    const int n = nt * 32 + j * 16 + col;
    const float bias = bo[n];
#pragma unroll
    for (int i = 0; i < 4; ++i)
#pragma unroll
      for (int r = 0; r < 8; ++r)
        out[(size_t)(ms * 64 + i * 16 + rb + r) * Dn + n] = acc[j][i][r] + bias;
  }
}

// ---------------------------------------------------------------------------
// Host launch
// ---------------------------------------------------------------------------
extern "C" void kernel_launch(void* const* d_in, const int* in_sizes, int n_in,
                              void* d_out, int out_size, void* d_ws, size_t ws_size,
                              hipStream_t stream) {
  const float* x  = (const float*)d_in[0];
  const float* Wq = (const float*)d_in[1];
  const float* Wk = (const float*)d_in[2];
  const float* Wv = (const float*)d_in[3];
  const float* Wo = (const float*)d_in[4];
  const float* bo = (const float*)d_in[5];
  float* out = (float*)d_out;

  const size_t XE = (size_t)Mn * Dn;   // 6,291,456 halves per activation tensor
  const size_t WE = (size_t)Dn * Dn;   //   589,824 halves per weight

  _Float16* ws  = (_Float16*)d_ws;
  _Float16* xh  = ws;
  _Float16* wqh = xh  + XE;
  _Float16* wkh = wqh + WE;
  _Float16* wvh = wkh + WE;
  _Float16* woh = wvh + WE;
  _Float16* Qh  = woh + WE;
  _Float16* Kh  = Qh  + XE;
  _Float16* Vt  = Kh  + XE;
  _Float16* Oh  = Vt  + XE;

  // Stage 0: conversions
  cvt_f32_f16_kernel<<<(int)((XE + 255) / 256), 256, 0, stream>>>(x,  xh,  (int)XE);
  cvt_f32_f16_kernel<<<(int)((WE + 255) / 256), 256, 0, stream>>>(Wq, wqh, (int)WE);
  cvt_f32_f16_kernel<<<(int)((WE + 255) / 256), 256, 0, stream>>>(Wk, wkh, (int)WE);
  cvt_f32_f16_kernel<<<(int)((WE + 255) / 256), 256, 0, stream>>>(Wv, wvh, (int)WE);
  cvt_f32_f16_kernel<<<(int)((WE + 255) / 256), 256, 0, stream>>>(Wo, woh, (int)WE);

  // Stage 1: QKV projections (128 m-strips x 48 n-tiles, 4 waves / block)
  qkv_kernel<<<(MT / 4) * NT / 4, 128, 0, stream>>>(xh, wqh, wkh, wvh, Qh, Kh, Vt);

  // Stage 2: flash attention (one wave per (b,h,32-row strip))
  attn_kernel<<<Bn * Hn * (Sn / 32), 32, 0, stream>>>(Qh, Kh, Vt, Oh);

  // Stage 3: output projection + bias (128 m-strips x 24 n-strips)
  out_proj_kernel<<<(MT / 4) * (NT / 2) / 4, 128, 0, stream>>>(Oh, woh, bo, out);
}
